// SparseAutoencoder_60266981097959
// MI455X (gfx1250) — compile-verified, matrix operand order checked by reading, not run on hardware
//
#include <hip/hip_runtime.h>

#define BB 8192
#define DD 768
#define LL 16384
#define KK 100

typedef __attribute__((ext_vector_type(16))) __bf16 v16bf;
typedef __attribute__((ext_vector_type(8)))  float  v8f;

static __device__ __forceinline__ unsigned short f2bf(float f) {
    unsigned int u = __float_as_uint(f);
    unsigned int r = (u + 0x7FFFu + ((u >> 16) & 1u)) >> 16;   // RNE
    return (unsigned short)r;
}
static __device__ __forceinline__ unsigned int f2key(float f) {
    unsigned int u = __float_as_uint(f);
    return (u & 0x80000000u) ? ~u : (u | 0x80000000u);         // monotonic key
}
static __device__ __forceinline__ float key2f(unsigned int k) {
    unsigned int u = (k & 0x80000000u) ? (k & 0x7FFFFFFFu) : ~k;
    return __uint_as_float(u);
}

// async DMA: global -> LDS, 32 bytes per call-site pair (b128 + b128 @ offset 16)
// GVS addressing: SGPR-pair base + 32-bit per-lane voffset (saves VGPR pairs).
// INST_OFFSET applies to BOTH the global and the LDS address (ISA 08_async_tensor §4.4)
static __device__ __forceinline__ void async_copy_32B(unsigned lds_addr, unsigned voff,
                                                      const void* sbase) {
    asm volatile("global_load_async_to_lds_b128 %0, %1, %2"
                 :: "v"(lds_addr), "v"(voff), "s"(sbase) : "memory");
    asm volatile("global_load_async_to_lds_b128 %0, %1, %2 offset:16"
                 :: "v"(lds_addr), "v"(voff), "s"(sbase) : "memory");
}
static __device__ __forceinline__ void wait_async0() {
    asm volatile("s_wait_asynccnt 0x0" ::: "memory");
}

// ---------------------------------------------------------------------------
// 1) W_enc [D,L] f32  ->  W_encT [L,D] bf16 (tiled transpose through LDS)
// ---------------------------------------------------------------------------
__global__ __launch_bounds__(1024) void wenc_bf16T_kernel(
    const float* __restrict__ W, unsigned short* __restrict__ Wt)
{
    __shared__ float tile[32][33];
    const int n0 = blockIdx.x * 32, d0 = blockIdx.y * 32;
    const int tx = threadIdx.x, ty = threadIdx.y;
    tile[ty][tx] = W[(size_t)(d0 + ty) * LL + n0 + tx];
    __syncthreads();
    Wt[(size_t)(n0 + ty) * DD + d0 + tx] = f2bf(tile[tx][ty]);
}

// ---------------------------------------------------------------------------
// 2) per-row normalize: xnc = (x-mu)/(std+eps) - pre_bias   (bf16), stats
// ---------------------------------------------------------------------------
__global__ __launch_bounds__(256) void normalize_kernel(
    const float* __restrict__ x, const float* __restrict__ pre_bias,
    unsigned short* __restrict__ xnc, float2* __restrict__ stats)
{
    __shared__ float2 red[256];
    const int b = blockIdx.x, tid = threadIdx.x;
    const float* row = x + (size_t)b * DD;
    float v[3], s = 0.f, s2 = 0.f;
    #pragma unroll
    for (int j = 0; j < 3; ++j) {
        v[j] = row[tid + j * 256];
        s += v[j]; s2 += v[j] * v[j];
    }
    red[tid] = make_float2(s, s2);
    __syncthreads();
    for (int off = 128; off > 0; off >>= 1) {
        if (tid < off) {
            red[tid].x += red[tid + off].x;
            red[tid].y += red[tid + off].y;
        }
        __syncthreads();
    }
    const float mu  = red[0].x * (1.f / DD);
    const float var = fmaxf(red[0].y * (1.f / DD) - mu * mu, 0.f);
    const float sd  = sqrtf(var);
    const float inv = 1.f / (sd + 1e-5f);
    if (tid == 0) stats[b] = make_float2(mu, sd);
    #pragma unroll
    for (int j = 0; j < 3; ++j) {
        const int d = tid + j * 256;
        xnc[(size_t)b * DD + d] = f2bf((v[j] - mu) * inv - pre_bias[d]);
    }
}

// ---------------------------------------------------------------------------
// 3) encode GEMM: C[B,L] = A_bf16[B,D] * WencT_bf16[L,D]^T + latent_bias
//    128x128 block tile, K-step 32, double-buffered LDS,
//    tile feed via GLOBAL_LOAD_ASYNC_TO_LDS_B128 (GVS mode, ASYNCcnt),
//    compute via v_wmma_f32_16x16x32_bf16. Fully unrolled k-loop (KT=24).
//    8 waves: 4 (M) x 2 (N); wave tile 32x64 = 2x4 wmma tiles
// ---------------------------------------------------------------------------
__global__ __launch_bounds__(256) void encode_gemm_kernel(
    const unsigned short* __restrict__ A,   // [B,D] bf16
    const unsigned short* __restrict__ Bt,  // [L,D] bf16
    const float* __restrict__ latent_bias,  // [L]
    float* __restrict__ C)                  // [B,L]
{
    constexpr int BK = 32;
    constexpr int KT = DD / BK;             // 24
    __shared__ __align__(16) unsigned short As[2][128][BK];
    __shared__ __align__(16) unsigned short Bs[2][128][BK];

    const int tid  = threadIdx.x;
    const int lane = tid & 31;
    const int half = lane >> 4;
    const int idx  = lane & 15;
    const int wave = tid >> 5;
    const int wm   = wave & 3;              // M-wave: offset 32*wm
    const int wn   = wave >> 2;             // N-wave: offset 64*wn

    const int m0 = blockIdx.y * 128;
    const int n0 = blockIdx.x * 128;

    // tile loader: 256 threads, each thread DMAs 32B of A and 32B of B per tile
    const int lr = tid >> 1;
    const int lc = (tid & 1) << 4;
    // per-lane byte offsets (i32 VADDR for GVS mode); bases stay in SGPR pairs
    const unsigned voffA = (unsigned)(((m0 + lr) * DD + lc) * 2);
    const unsigned voffB = (unsigned)(((n0 + lr) * DD + lc) * 2);

    // LDS byte addresses for this thread's slice (addr[31:0] == LDS offset)
    const unsigned ldsA0 = (unsigned)(uintptr_t)&As[0][lr][lc];
    const unsigned ldsA1 = (unsigned)(uintptr_t)&As[1][lr][lc];
    const unsigned ldsB0 = (unsigned)(uintptr_t)&Bs[0][lr][lc];
    const unsigned ldsB1 = (unsigned)(uintptr_t)&Bs[1][lr][lc];

    v8f acc[2][4] = {};

    // preload k-tile 0 via async DMA
    async_copy_32B(ldsA0, voffA, A);
    async_copy_32B(ldsB0, voffB, Bt);
    wait_async0();
    __syncthreads();

    #pragma unroll
    for (int kt = 0; kt < KT; ++kt) {
        const int cur = kt & 1;

        // kick off DMA of the next k-tile into the other buffer
        if (kt + 1 < KT) {
            const unsigned kb = (unsigned)((kt + 1) * BK * 2);   // 64 B per k-step
            async_copy_32B(cur ? ldsA0 : ldsA1, voffA + kb, A);
            async_copy_32B(cur ? ldsB0 : ldsB1, voffB + kb, Bt);
        }

        // A fragments: lane holds M=idx, K split by half per ISA 16-bit layout
        v16bf af[2], bfv[4];
        #pragma unroll
        for (int mt = 0; mt < 2; ++mt) {
            const unsigned short* p = &As[cur][wm * 32 + mt * 16 + idx][half * 8];
            *((uint4*)&af[mt])      = *(const uint4*)(p);
            *((uint4*)&af[mt] + 1)  = *(const uint4*)(p + 16);
        }
        #pragma unroll
        for (int nt = 0; nt < 4; ++nt) {
            const unsigned short* p = &Bs[cur][wn * 64 + nt * 16 + idx][half * 8];
            *((uint4*)&bfv[nt])     = *(const uint4*)(p);
            *((uint4*)&bfv[nt] + 1) = *(const uint4*)(p + 16);
        }

        #pragma unroll
        for (int mt = 0; mt < 2; ++mt)
            #pragma unroll
            for (int nt = 0; nt < 4; ++nt)
                acc[mt][nt] = __builtin_amdgcn_wmma_f32_16x16x32_bf16(
                    false, af[mt], false, bfv[nt],
                    (short)0, acc[mt][nt], false, false);

        // DMA of next tile must land before anyone reads it
        wait_async0();
        __syncthreads();
    }

    // epilogue: C/D layout — VGPR i holds M=i (lanes 0-15) / M=8+i (lanes 16-31)
    #pragma unroll
    for (int mt = 0; mt < 2; ++mt)
        #pragma unroll
        for (int nt = 0; nt < 4; ++nt) {
            const int gn = n0 + wn * 64 + nt * 16 + idx;
            const float bias = latent_bias[gn];
            const int gmBase = m0 + wm * 32 + mt * 16 + half * 8;
            #pragma unroll
            for (int i = 0; i < 8; ++i)
                C[(size_t)(gmBase + i) * LL + gn] = acc[mt][nt][i] + bias;
        }
}

// ---------------------------------------------------------------------------
// 4) per-row exact top-K: 4-pass 8-bit radix select, keys cached in LDS
// ---------------------------------------------------------------------------
__global__ __launch_bounds__(256) void topk_kernel(
    const float* __restrict__ pre,   // [B,L] latents_pre_act
    float* __restrict__ lat,         // [B,L] sparse latents
    int* __restrict__ selIdx, float* __restrict__ selVal,
    int* __restrict__ selCnt)
{
    extern __shared__ unsigned int smem[];
    unsigned int* keys = smem;           // LL entries (64 KB)
    unsigned int* hist = smem + LL;      // 256 entries
    __shared__ unsigned int s_prefix;
    __shared__ int s_need, s_tie, s_cnt;

    const int b = blockIdx.x, tid = threadIdx.x;
    const float* in  = pre + (size_t)b * LL;
    float*       out = lat + (size_t)b * LL;

    for (int j = 0; j < LL / 256; ++j) {
        const int i = tid + j * 256;
        keys[i] = f2key(in[i]);
    }
    if (tid == 0) { s_prefix = 0; s_need = KK; s_tie = 0; s_cnt = 0; }
    __syncthreads();

    for (int pass = 0; pass < 4; ++pass) {
        hist[tid] = 0;
        __syncthreads();
        const int shift = 24 - 8 * pass;
        const unsigned int pref = s_prefix;
        for (int j = 0; j < LL / 256; ++j) {
            const unsigned int k = keys[tid + j * 256];
            const bool cand = (pass == 0) || ((k >> (shift + 8)) == pref);
            if (cand) atomicAdd(&hist[(k >> shift) & 255u], 1u);
        }
        __syncthreads();
        if (tid == 0) {
            const int need = s_need;
            unsigned int cum = 0; int bin = 0;
            for (int bb = 255; bb >= 0; --bb) {
                const unsigned int c = hist[bb];
                if (cum + c >= (unsigned int)need) { bin = bb; s_need = need - (int)cum; break; }
                cum += c;
            }
            s_prefix = (pref << 8) | (unsigned int)bin;
        }
        __syncthreads();
    }

    const unsigned int thresh = s_prefix;
    const int need = s_need;
    __syncthreads();

    for (int j = 0; j < LL / 256; ++j) {
        const int i = tid + j * 256;
        const unsigned int k = keys[i];
        bool keep = (k > thresh);
        if (!keep && k == thresh) {
            const int old = atomicAdd(&s_tie, 1);
            keep = (old < need);
        }
        const float v = key2f(k);
        out[i] = keep ? v : 0.0f;
        if (keep) {
            const int s = atomicAdd(&s_cnt, 1);
            if (s < KK) { selIdx[b * KK + s] = i; selVal[b * KK + s] = v; }
        }
    }
    __syncthreads();
    if (tid == 0) selCnt[b] = (s_cnt < KK) ? s_cnt : KK;
}

// ---------------------------------------------------------------------------
// 5) sparse decode: recons = (sum_k val*W_dec[idx] + pre_bias) * std + mu
// ---------------------------------------------------------------------------
__global__ __launch_bounds__(256) void decode_kernel(
    const float* __restrict__ Wdec, const float* __restrict__ pre_bias,
    const int* __restrict__ selIdx, const float* __restrict__ selVal,
    const int* __restrict__ selCnt, const float2* __restrict__ stats,
    float* __restrict__ out)
{
    __shared__ float sval[KK];
    __shared__ int   sidx[KK];
    const int b = blockIdx.x, tid = threadIdx.x;
    const int cnt = selCnt[b];
    if (tid < KK) {
        sval[tid] = (tid < cnt) ? selVal[b * KK + tid] : 0.f;
        sidx[tid] = (tid < cnt) ? selIdx[b * KK + tid] : 0;
    }
    __syncthreads();
    const float2 st = stats[b];
    #pragma unroll
    for (int j = 0; j < 3; ++j) {
        const int d = tid + j * 256;
        float acc = pre_bias[d];
        #pragma unroll 4
        for (int t = 0; t < KK; ++t)
            acc = fmaf(sval[t], Wdec[(size_t)sidx[t] * DD + d], acc);
        out[(size_t)b * DD + d] = acc * st.y + st.x;
    }
}

// ---------------------------------------------------------------------------
extern "C" void kernel_launch(void* const* d_in, const int* in_sizes, int n_in,
                              void* d_out, int out_size, void* d_ws, size_t ws_size,
                              hipStream_t stream)
{
    (void)in_sizes; (void)n_in; (void)out_size; (void)ws_size;
    const float* x           = (const float*)d_in[0];
    const float* pre_bias    = (const float*)d_in[1];
    const float* latent_bias = (const float*)d_in[2];
    const float* W_enc       = (const float*)d_in[3];
    const float* W_dec       = (const float*)d_in[4];

    float* out0 = (float*)d_out;                          // latents_pre_act [B,L]
    float* out1 = out0 + (size_t)BB * LL;                 // latents         [B,L]
    float* out2 = out1 + (size_t)BB * LL;                 // recons          [B,D]

    char* ws = (char*)d_ws;
    size_t off = 0;
    auto carve = [&](size_t bytes) { size_t o = off; off = (off + bytes + 255) & ~(size_t)255; return o; };
    unsigned short* xnc   = (unsigned short*)(ws + carve((size_t)BB * DD * 2));
    unsigned short* wencT = (unsigned short*)(ws + carve((size_t)LL * DD * 2));
    float2*         stats = (float2*)       (ws + carve((size_t)BB * sizeof(float2)));
    int*            sIdx  = (int*)          (ws + carve((size_t)BB * KK * 4));
    float*          sVal  = (float*)        (ws + carve((size_t)BB * KK * 4));
    int*            sCnt  = (int*)          (ws + carve((size_t)BB * 4));

    wenc_bf16T_kernel<<<dim3(LL / 32, DD / 32), dim3(32, 32), 0, stream>>>(W_enc, wencT);
    normalize_kernel<<<BB, 256, 0, stream>>>(x, pre_bias, xnc, stats);
    encode_gemm_kernel<<<dim3(LL / 128, BB / 128), 256, 0, stream>>>(xnc, wencT, latent_bias, out0);
    topk_kernel<<<BB, 256, (LL + 256) * sizeof(unsigned int), stream>>>(out0, out1, sIdx, sVal, sCnt);
    decode_kernel<<<BB, 256, 0, stream>>>(W_dec, pre_bias, sIdx, sVal, sCnt, stats, out2);
}
